// LightweightSelfAttention_51694226375319
// MI455X (gfx1250) — matrix operand order repaired and checked by reading, not compiled, not run on hardware
//
#include <hip/hip_runtime.h>
#include <hip/hip_bf16.h>
#include <stddef.h>

// ---------------------------------------------------------------------------
// LightweightSelfAttention for MI455X (gfx1250), wave32 + WMMA bf16.
// B=4, C=256, N=4096, Cr=32.
// ---------------------------------------------------------------------------

typedef __attribute__((ext_vector_type(16))) __bf16 v16bf;
typedef __attribute__((ext_vector_type(8)))  __bf16 v8bf;
typedef __attribute__((ext_vector_type(8)))  float  v8f;

#define NB   4
#define NC   256
#define NN   4096
#define NCR  32

// workspace layout (bf16 elements)
#define OFF_WQ 0u                          // [32][256]
#define OFF_WK 8192u                       // [32][256]
#define OFF_WV 16384u                      // [256][256]
#define OFF_Q  81920u                      // [B][N][32]   q[n][o]
#define OFF_K  606208u                     // [B][N][32]   k^T[n][o]
#define OFF_V  1130496u                    // [B][N/16][C][16]  vblk[nb][c][n%16]

static __device__ inline v8f vzero8() {
  v8f z;
#pragma unroll
  for (int i = 0; i < 8; ++i) z[i] = 0.0f;
  return z;
}

static __device__ inline v8f wmma_bf16(v16bf a, v16bf b, v8f c) {
  return __builtin_amdgcn_wmma_f32_16x16x32_bf16(false, a, false, b,
                                                 (short)0, c, false, false);
}

// ---------------------------------------------------------------------------
// Kernel 1: weights f32 -> bf16
// ---------------------------------------------------------------------------
__global__ void lsa_prep(const float* __restrict__ Wq,
                         const float* __restrict__ Wk,
                         const float* __restrict__ Wv,
                         __bf16* __restrict__ ws) {
  int i = blockIdx.x * 256 + threadIdx.x;        // 65536 threads
  if (i < 8192) {
    ws[OFF_WQ + i] = (__bf16)Wq[i];
    ws[OFF_WK + i] = (__bf16)Wk[i];
  }
  if (i < 65536) ws[OFF_WV + i] = (__bf16)Wv[i];
}

// ---------------------------------------------------------------------------
// Kernel 2: projections q,k,v via WMMA.
// One wave per (b, 16-column block of N). A = x^T tile (16n x 32c),
// B = W^T tile (32c x 16o). D tiles land as [n][o] -> fragment-ready stores.
// ---------------------------------------------------------------------------
__global__ __launch_bounds__(256) void lsa_proj(const float* __restrict__ x,
                                                __bf16* __restrict__ ws) {
  const int lane = threadIdx.x & 31;
  const int wave = threadIdx.x >> 5;
  const int tile = blockIdx.x * 8 + wave;        // 1024 tiles
  const int b  = tile >> 8;
  const int nb = tile & 255;
  const int n0 = nb * 16;
  const int r  = lane & 15;                      // A row / D col
  const int g  = lane >> 4;
  const int abase = 8 * g;                       // unused (A also uses kbase)
  const int kbase = 16 * g;                      // B k-chunk base

  const __bf16* Wqb = ws + OFF_WQ;
  const __bf16* Wkb = ws + OFF_WK;
  const __bf16* Wvb = ws + OFF_WV;

  v8f accq[2], acck[2], accv[16];
#pragma unroll
  for (int t = 0; t < 2; ++t) { accq[t] = vzero8(); acck[t] = vzero8(); }
#pragma unroll
  for (int t = 0; t < 16; ++t) accv[t] = vzero8();

  const float* xcol = x + (size_t)b * NC * NN + n0 + r;   // column n0+r of x

  for (int kk = 0; kk < 8; ++kk) {               // K loop over C in steps of 32
    const int c0 = kk * 32;
    // A fragment: x^T[r][c0 .. c0+32), contiguous k order (k = 16g + e)
    v16bf a;
#pragma unroll
    for (int e = 0; e < 8; ++e)
      a[e] = (__bf16)xcol[(size_t)(c0 + kbase + e) * NN];
#pragma unroll
    for (int e = 0; e < 8; ++e)
      a[8 + e] = (__bf16)xcol[(size_t)(c0 + kbase + 8 + e) * NN];

#pragma unroll
    for (int t = 0; t < 2; ++t) {
      const int o = t * 16 + r;                  // B column
      v16bf bq = *(const v16bf*)(Wqb + (size_t)o * NC + c0 + kbase);
      v16bf bk = *(const v16bf*)(Wkb + (size_t)o * NC + c0 + kbase);
      accq[t] = wmma_bf16(a, bq, accq[t]);
      acck[t] = wmma_bf16(a, bk, acck[t]);
    }
#pragma unroll
    for (int t = 0; t < 16; ++t) {
      const int o = t * 16 + r;
      v16bf bv = *(const v16bf*)(Wvb + (size_t)o * NC + c0 + kbase);
      accv[t] = wmma_bf16(a, bv, accv[t]);
    }
  }

  // store q^T / k^T rows: qb[b][n][o]
  __bf16* qb = ws + OFF_Q + (size_t)b * NN * NCR;
  __bf16* kb = ws + OFF_K + (size_t)b * NN * NCR;
#pragma unroll
  for (int t = 0; t < 2; ++t)
#pragma unroll
    for (int d = 0; d < 8; ++d) {
      const int n = n0 + d + 8 * g;              // D row
      const int o = t * 16 + r;                  // D col
      qb[(size_t)n * NCR + o] = (__bf16)accq[t][d];
      kb[(size_t)n * NCR + o] = (__bf16)acck[t][d];
    }
  // store v in blocked fragment layout: vblk[b][nb][c][n%16]
  __bf16* vb = ws + OFF_V + ((size_t)b * 256 + nb) * NC * 16;
#pragma unroll
  for (int t = 0; t < 16; ++t)
#pragma unroll
    for (int d = 0; d < 8; ++d) {
      const int c  = t * 16 + r;
      const int nl = d + 8 * g;
      vb[(size_t)c * 16 + nl] = (__bf16)accv[t][d];
    }
}

// ---------------------------------------------------------------------------
// Kernel 3: flash attention. One wave per 16-query tile, 32 keys / iter.
// ---------------------------------------------------------------------------
__global__ __launch_bounds__(256) void lsa_attn(const float* __restrict__ x,
                                                const __bf16* __restrict__ ws,
                                                float* __restrict__ out) {
  __shared__ __align__(32) char smem[8 * 1024];  // 1KB private per wave

  const int lane = threadIdx.x & 31;
  const int wave = threadIdx.x >> 5;
  const int tile = blockIdx.x * 8 + wave;        // 1024 tiles
  const int b  = tile >> 8;
  const int nb = tile & 255;
  const int n0 = nb * 16;
  const int r  = lane & 15;
  const int g  = lane >> 4;
  const int abase = 8 * g;
  const int kbase = 16 * g;

  __bf16* sP = (__bf16*)(smem + wave * 1024);    // [16][32] bf16 P staging
  float*  sT = (float*)(smem + wave * 1024);     // [16][16] f32 O transpose

  const __bf16* qb = ws + OFF_Q + (size_t)b * NN * NCR;
  const __bf16* kb = ws + OFF_K + (size_t)b * NN * NCR;
  const __bf16* vb = ws + OFF_V + (size_t)b * 256 * NC * 16;

  // Q fragment (row n0+r, split into the two 8-element K chunks per lane)
  const __bf16* qrow = qb + (size_t)(n0 + r) * NCR;
  v16bf aq;
  {
    v8bf q1 = *(const v8bf*)(qrow + kbase);
    v8bf q2 = *(const v8bf*)(qrow + kbase + 8);
#pragma unroll
    for (int e = 0; e < 8; ++e) { aq[e] = q1[e]; aq[8 + e] = q2[e]; }
  }

  float m[8], l[8];
  v8f O[16];
#pragma unroll
  for (int d = 0; d < 8; ++d) { m[d] = -3.0e38f; l[d] = 0.0f; }
#pragma unroll
  for (int t = 0; t < 16; ++t) O[t] = vzero8();

  const float L2E = 1.4426950408889634f;

  for (int kt = 0; kt < 128; ++kt) {
    const int kn0 = kt * 32;

    // S = Q * K^T : two 16x16 tiles, K-dim = 32 (head dim)
    v16bf bk0 = *(const v16bf*)(kb + (size_t)(kn0 + r) * NCR + kbase);
    v16bf bk1 = *(const v16bf*)(kb + (size_t)(kn0 + 16 + r) * NCR + kbase);
    if (kt < 127)  // prefetch next K block (global_prefetch on gfx1250)
      __builtin_prefetch(kb + (size_t)(kn0 + 32 + r) * NCR, 0, 1);
    v8f s0 = wmma_bf16(aq, bk0, vzero8());
    v8f s1 = wmma_bf16(aq, bk1, vzero8());

    // online softmax: row i = d + 8g lives across the 16 lanes of this half
    float corr[8];
#pragma unroll
    for (int d = 0; d < 8; ++d) {
      float bm = fmaxf(s0[d], s1[d]);
      bm = fmaxf(bm, __shfl_xor(bm, 1, 32));
      bm = fmaxf(bm, __shfl_xor(bm, 2, 32));
      bm = fmaxf(bm, __shfl_xor(bm, 4, 32));
      bm = fmaxf(bm, __shfl_xor(bm, 8, 32));
      const float mn = fmaxf(m[d], bm);
      corr[d] = __builtin_amdgcn_exp2f((m[d] - mn) * L2E);
      const float p0 = __builtin_amdgcn_exp2f((s0[d] - mn) * L2E);
      const float p1 = __builtin_amdgcn_exp2f((s1[d] - mn) * L2E);
      float bs = p0 + p1;
      bs += __shfl_xor(bs, 1, 32);
      bs += __shfl_xor(bs, 2, 32);
      bs += __shfl_xor(bs, 4, 32);
      bs += __shfl_xor(bs, 8, 32);
      l[d] = l[d] * corr[d] + bs;
      m[d] = mn;
      s0[d] = p0; s1[d] = p1;                    // reuse as P
    }
    // rescale accumulators
#pragma unroll
    for (int t = 0; t < 16; ++t)
#pragma unroll
      for (int d = 0; d < 8; ++d) O[t][d] *= corr[d];

    // P: D-layout -> A-layout through private LDS
    __builtin_amdgcn_wave_barrier();
#pragma unroll
    for (int d = 0; d < 8; ++d) {
      const int i = d + 8 * g;
      sP[i * 32 + r]      = (__bf16)s0[d];
      sP[i * 32 + 16 + r] = (__bf16)s1[d];
    }
    __builtin_amdgcn_wave_barrier();
    v16bf pa;
    {
      v8bf p1v = *(const v8bf*)(sP + r * 32 + kbase);
      v8bf p2v = *(const v8bf*)(sP + r * 32 + kbase + 8);
#pragma unroll
      for (int e = 0; e < 8; ++e) { pa[e] = p1v[e]; pa[8 + e] = p2v[e]; }
    }

    // O += P * V^T : 16 c-tiles, V fragments contiguous per lane
    const __bf16* vblkp = vb + (size_t)(kn0 / 16 + g) * NC * 16;
#pragma unroll
    for (int t = 0; t < 16; ++t) {
      v16bf bv = *(const v16bf*)(vblkp + (size_t)(t * 16 + r) * 16);
      O[t] = wmma_bf16(pa, bv, O[t]);
    }
  }

  // normalize by row sums
  float rl[8];
#pragma unroll
  for (int d = 0; d < 8; ++d) rl[d] = 1.0f / l[d];
#pragma unroll
  for (int t = 0; t < 16; ++t)
#pragma unroll
    for (int d = 0; d < 8; ++d) O[t][d] *= rl[d];

  // transpose each 16x16 tile through LDS, add residual, coalesced store
  const float* xb = x   + (size_t)b * NC * NN;
  float*       ob = out + (size_t)b * NC * NN;
  for (int t = 0; t < 16; ++t) {
    __builtin_amdgcn_wave_barrier();
#pragma unroll
    for (int d = 0; d < 8; ++d) sT[(d + 8 * g) * 16 + r] = O[t][d];
    __builtin_amdgcn_wave_barrier();
#pragma unroll
    for (int d = 0; d < 8; ++d) {
      const int c = t * 16 + d + 8 * g;
      const size_t idx = (size_t)c * NN + n0 + r;
      ob[idx] = sT[r * 16 + d + 8 * g] + xb[idx];
    }
  }
}

// ---------------------------------------------------------------------------
extern "C" void kernel_launch(void* const* d_in, const int* in_sizes, int n_in,
                              void* d_out, int out_size, void* d_ws, size_t ws_size,
                              hipStream_t stream) {
  (void)in_sizes; (void)n_in; (void)out_size; (void)ws_size;
  const float* x  = (const float*)d_in[0];
  const float* Wq = (const float*)d_in[1];
  const float* Wk = (const float*)d_in[2];
  const float* Wv = (const float*)d_in[3];
  float* out = (float*)d_out;
  __bf16* ws = (__bf16*)d_ws;

  lsa_prep<<<256, 256, 0, stream>>>(Wq, Wk, Wv, ws);
  lsa_proj<<<128, 256, 0, stream>>>(x, ws);
  lsa_attn<<<128, 256, 0, stream>>>(x, ws, out);
}
